// Transformer_31817117728961
// MI455X (gfx1250) — matrix operand order repaired and checked by reading, not compiled, not run on hardware
//
#include <hip/hip_runtime.h>

// Point-transformer layer for MI455X (gfx1250, wave32).
// Bandwidth-bound: gather of fk/fv dominates -> store them f16 in workspace.
// WMMA used for (1) QKV projections, (2) per-point 16x64 @ 64x8 attention GEMM.

#define NPTS 100000
#define CIN  64
#define COUT 64
#define NSN  16
#define HH   8          // COUT / share_planes
#define EPSV 1e-5f

typedef _Float16 f16;
typedef __attribute__((ext_vector_type(16))) _Float16 v16h;
typedef __attribute__((ext_vector_type(8)))  _Float16 v8h;
typedef __attribute__((ext_vector_type(8)))  float    v8f;

#define WMMA_F16(A, B, C) \
    __builtin_amdgcn_wmma_f32_16x16x32_f16(false, (A), false, (B), (short)0, (C), false, false)

// ---- workspace layout (bytes) ----
#define OFF_FQ    0ull                         // N*64 f32  = 25,600,000
#define OFF_FKH   25600000ull                  // N*64 f16  = 12,800,000
#define OFF_FVH   38400000ull                  // N*64 f16  = 12,800,000
#define OFF_BNP   51200000ull                  // 160 f32 folded-BN params
#define OFF_BQKV  51201024ull                  // 768*16 f16 B-fragments (Wq/Wk/Wv)
#define OFF_BW1   51225600ull                  // 64*16  f16 B-fragments (w_W1 padded)

// ------------------------------------------------------------------
// Kernel 0: fold BN params, pre-pack weight matrices into WMMA
// B-fragment layout (lane = column, lane-half selects K-half).
// ------------------------------------------------------------------
__global__ void prep_kernel(const float* __restrict__ Wq, const float* __restrict__ Wk,
                            const float* __restrict__ Wv,
                            const float* __restrict__ pg1, const float* __restrict__ pbe1,
                            const float* __restrict__ pm1, const float* __restrict__ pv1,
                            const float* __restrict__ wg1, const float* __restrict__ wbe1,
                            const float* __restrict__ wm1, const float* __restrict__ wv1,
                            const float* __restrict__ wg2, const float* __restrict__ wbe2,
                            const float* __restrict__ wm2, const float* __restrict__ wv2,
                            const float* __restrict__ wW1,
                            float* __restrict__ bnp, f16* __restrict__ bqkv, f16* __restrict__ bw1)
{
    const int e = blockIdx.x * blockDim.x + threadIdx.x;
    if (e < 768) {
        // entry = ((mat*2 + kb)*4 + nt)*32 + lane ; 16 halfs each
        const int lane = e & 31;
        const int rest = e >> 5;
        const int nt  = rest & 3;
        const int kb  = (rest >> 2) & 1;
        const int mat = rest >> 3;
        const float* W = (mat == 0) ? Wq : (mat == 1) ? Wk : Wv;
        const int col  = nt * 16 + (lane & 15);
        const int half = lane >> 4;
#pragma unroll
        for (int j = 0; j < 16; ++j) {
            const int k = kb * 32 + (j >> 3) * 16 + half * 8 + (j & 7);
            bqkv[e * 16 + j] = (f16)W[k * COUT + col];
        }
    } else if (e < 832) {
        // w_W1 (64x8) padded to 16 columns, B-fragment layout
        const int e2   = e - 768;
        const int lane = e2 & 31;
        const int kb   = e2 >> 5;
        const int col  = lane & 15;
        const int half = lane >> 4;
#pragma unroll
        for (int j = 0; j < 16; ++j) {
            const int k = kb * 32 + (j >> 3) * 16 + half * 8 + (j & 7);
            bw1[e2 * 16 + j] = (col < HH) ? (f16)wW1[k * HH + col] : (f16)0.f;
        }
    } else if (e < 832 + 64) {
        const int t = e - 832;
        if (t < 3) {
            float s = pg1[t] * rsqrtf(pv1[t] + EPSV);
            bnp[t] = s; bnp[4 + t] = pbe1[t] - pm1[t] * s;
        }
        { // t < 64 always here
            float s = wg1[t] * rsqrtf(wv1[t] + EPSV);
            bnp[8 + t] = s; bnp[72 + t] = wbe1[t] - wm1[t] * s;
        }
        if (t < HH) {
            float s = wg2[t] * rsqrtf(wv2[t] + EPSV);
            bnp[136 + t] = s; bnp[144 + t] = wbe2[t] - wm2[t] * s;
        }
    }
}

// ------------------------------------------------------------------
// Kernel 1: fq/fk/fv = feat @ W + b via v_wmma_f32_16x16x32_f16.
// One wave per 16-row tile (N = 6250 * 16 exactly). fk/fv stored f16.
// ------------------------------------------------------------------
__global__ void qkv_kernel(const float* __restrict__ feat,
                           const float* __restrict__ bq, const float* __restrict__ bk,
                           const float* __restrict__ bv,
                           const f16* __restrict__ Bpacked,
                           float* __restrict__ fq, f16* __restrict__ fkh, f16* __restrict__ fvh)
{
    const int lane = threadIdx.x & 31;
    const int wave = threadIdx.x >> 5;
    const int tile = blockIdx.x * (blockDim.x >> 5) + wave;
    if (tile >= NPTS / 16) return;                 // wave-uniform: EXEC stays full
    const int m = lane & 15, half = lane >> 4;
    const int row = tile * 16 + m;

    // A fragments: 16-bit A 16x32 layout. lane half selects K-groups
    // {half*8..+7, 16+half*8..+7} per 32-wide K block.
    v16h a[2];
#pragma unroll
    for (int kb = 0; kb < 2; ++kb) {
        const float* src = feat + row * CIN + kb * 32 + half * 8;
        const float4 x0 = *(const float4*)(src);
        const float4 x1 = *(const float4*)(src + 4);
        const float4 x2 = *(const float4*)(src + 16);
        const float4 x3 = *(const float4*)(src + 20);
        v16h av;
        av[0] = (f16)x0.x; av[1] = (f16)x0.y; av[2]  = (f16)x0.z; av[3]  = (f16)x0.w;
        av[4] = (f16)x1.x; av[5] = (f16)x1.y; av[6]  = (f16)x1.z; av[7]  = (f16)x1.w;
        av[8] = (f16)x2.x; av[9] = (f16)x2.y; av[10] = (f16)x2.z; av[11] = (f16)x2.w;
        av[12] = (f16)x3.x; av[13] = (f16)x3.y; av[14] = (f16)x3.z; av[15] = (f16)x3.w;
        a[kb] = av;
    }

    const v16h* B = (const v16h*)Bpacked;
#pragma unroll
    for (int mat = 0; mat < 3; ++mat) {
        const float* bias = (mat == 0) ? bq : (mat == 1) ? bk : bv;
#pragma unroll
        for (int nt = 0; nt < 4; ++nt) {
            v16h b0 = B[((mat * 2 + 0) * 4 + nt) * 32 + lane];
            v16h b1 = B[((mat * 2 + 1) * 4 + nt) * 32 + lane];
            v8f c = {};
            c = WMMA_F16(a[0], b0, c);
            c = WMMA_F16(a[1], b1, c);
            const int col = nt * 16 + m;
            const float bb = bias[col];
            if (mat == 0) {
#pragma unroll
                for (int r = 0; r < 8; ++r)
                    fq[(tile * 16 + half * 8 + r) * COUT + col] = c[r] + bb;
            } else {
                f16* dst = (mat == 1) ? fkh : fvh;
#pragma unroll
                for (int r = 0; r < 8; ++r)
                    dst[(tile * 16 + half * 8 + r) * COUT + col] = (f16)(c[r] + bb);
            }
        }
    }
}

// ------------------------------------------------------------------
// Kernel 2: one wave per point. Neighbor axis = WMMA M axis.
//   lanes: m = lane&15 -> neighbor, half -> channel block {0..7}/{8..15}
//   each lane owns channels {g*16 + half*8 + t : g=0..3, t=0..7}
// ------------------------------------------------------------------
__global__ void attn_kernel(const float* __restrict__ point, const int* __restrict__ idx,
                            const float* __restrict__ fq, const f16* __restrict__ fkh,
                            const f16* __restrict__ fvh,
                            const float* __restrict__ pW1, const float* __restrict__ pb1,
                            const float* __restrict__ pW2, const float* __restrict__ pb2,
                            const float* __restrict__ bnp, const f16* __restrict__ bw1,
                            const float* __restrict__ wb1,
                            const float* __restrict__ wW2, const float* __restrict__ wb2,
                            float* __restrict__ out)
{
    const int lane = threadIdx.x & 31;
    const int wave = threadIdx.x >> 5;
    const int i = blockIdx.x * (blockDim.x >> 5) + wave;   // exact: 12500*8 waves
    const int m = lane & 15, half = lane >> 4, chb = half * 8;

    __shared__ __align__(16) float xlds[8][16][HH];

    const int j = idx[i * NSN + m];                        // neighbor id for this lane
    __builtin_prefetch(idx + (i + 64) * NSN, 0, 0);        // global_prefetch_b8

    // relative coords
    const float pix = point[i * 3], piy = point[i * 3 + 1], piz = point[i * 3 + 2];
    const float g0 = point[j * 3] - pix;
    const float g1 = point[j * 3 + 1] - piy;
    const float g2 = point[j * 3 + 2] - piz;

    // pos-enc hidden: relu(bn(gx @ pW1 + pb1))
    float h0, h1, h2;
    {
        float hh[3];
#pragma unroll
        for (int d = 0; d < 3; ++d) {
            float t = g0 * pW1[d] + g1 * pW1[3 + d] + g2 * pW1[6 + d] + pb1[d];
            hh[d] = fmaxf(t * bnp[d] + bnp[4 + d], 0.f);
        }
        h0 = hh[0]; h1 = hh[1]; h2 = hh[2];
    }

    float aval[4][8];     // gv + pr for this lane's 32 channels
    v16h afrag[2];        // WMMA A fragments (neighbors x channels)
#pragma unroll
    for (int g = 0; g < 4; ++g) {
        const int cb = g * 16 + chb;
        const v8h gk = *(const v8h*)(fkh + (size_t)j * COUT + cb);   // b128 gather
        const v8h gv = *(const v8h*)(fvh + (size_t)j * COUT + cb);   // b128 gather
        const float4 q0 = *(const float4*)(fq + (size_t)i * COUT + cb);
        const float4 q1 = *(const float4*)(fq + (size_t)i * COUT + cb + 4);
        const float fqv[8] = {q0.x, q0.y, q0.z, q0.w, q1.x, q1.y, q1.z, q1.w};
#pragma unroll
        for (int t = 0; t < 8; ++t) {
            const int ch = cb + t;
            const float pr = h0 * pW2[ch] + h1 * pW2[COUT + ch] + h2 * pW2[2 * COUT + ch] + pb2[ch];
            const float wpre = (float)gk[t] - fqv[t] + pr;
            const float wrl = fmaxf(wpre * bnp[8 + ch] + bnp[72 + ch], 0.f);
            afrag[g >> 1][(g & 1) * 8 + t] = (f16)wrl;
            aval[g][t] = (float)gv[t] + pr;
        }
    }

    // 16x64 @ 64x8(padded 16): attention-weight projection
    const v16h* BW = (const v16h*)bw1;
    v16h b0 = BW[lane], b1 = BW[32 + lane];
    v8f d = {};
    d = WMMA_F16(afrag[0], b0, d);
    d = WMMA_F16(afrag[1], b1, d);

    // D layout: lane col = m (valid m<8), rows = neighbors half*8+r.
    // Apply bias+bn2+relu, transpose to LDS so every lane can see all 8 ch of its neighbor.
    if (m < HH) {
        const float sc = bnp[136 + m], sh = bnp[144 + m], bb = wb1[m];
#pragma unroll
        for (int r = 0; r < 8; ++r) {
            xlds[wave][half * 8 + r][m] = fmaxf((d[r] + bb) * sc + sh, 0.f);
        }
    }
    __builtin_amdgcn_wave_barrier();   // same-wave DS ops are HW-ordered; stop compiler reordering

    float xv[HH];
    {
        const float4 x0 = *(const float4*)&xlds[wave][m][0];
        const float4 x1 = *(const float4*)&xlds[wave][m][4];
        xv[0] = x0.x; xv[1] = x0.y; xv[2] = x0.z; xv[3] = x0.w;
        xv[4] = x1.x; xv[5] = x1.y; xv[6] = x1.z; xv[7] = x1.w;
    }

    // second linear 8->8 (per-neighbor), then softmax over the 16 neighbors
    float l[HH];
#pragma unroll
    for (int c2 = 0; c2 < HH; ++c2) {
        float acc = wb2[c2];
#pragma unroll
        for (int c = 0; c < HH; ++c) acc += xv[c] * wW2[c * HH + c2];
        l[c2] = acc;
    }
    float mx[HH];
#pragma unroll
    for (int c2 = 0; c2 < HH; ++c2) mx[c2] = l[c2];
#pragma unroll
    for (int off = 1; off < 16; off <<= 1)
#pragma unroll
        for (int c2 = 0; c2 < HH; ++c2) mx[c2] = fmaxf(mx[c2], __shfl_xor(mx[c2], off, 32));
    float ex[HH], sm[HH];
#pragma unroll
    for (int c2 = 0; c2 < HH; ++c2) { ex[c2] = __expf(l[c2] - mx[c2]); sm[c2] = ex[c2]; }
#pragma unroll
    for (int off = 1; off < 16; off <<= 1)
#pragma unroll
        for (int c2 = 0; c2 < HH; ++c2) sm[c2] += __shfl_xor(sm[c2], off, 32);

    // weighted sum over neighbors: out[ch] = sum_n (gv+pr)[n][ch] * attn[n][ch%8]
    float part[4][8];
#pragma unroll
    for (int g = 0; g < 4; ++g)
#pragma unroll
        for (int t = 0; t < 8; ++t) part[g][t] = aval[g][t] * (ex[t] / sm[t]);
#pragma unroll
    for (int off = 1; off < 16; off <<= 1)
#pragma unroll
        for (int g = 0; g < 4; ++g)
#pragma unroll
            for (int t = 0; t < 8; ++t) part[g][t] += __shfl_xor(part[g][t], off, 32);

    if (m == 0) {
#pragma unroll
        for (int g = 0; g < 4; ++g) {
            float4 o0 = {part[g][0], part[g][1], part[g][2], part[g][3]};
            float4 o1 = {part[g][4], part[g][5], part[g][6], part[g][7]};
            *(float4*)(out + (size_t)i * COUT + g * 16 + chb)     = o0;
            *(float4*)(out + (size_t)i * COUT + g * 16 + chb + 4) = o1;
        }
    }
}

// ------------------------------------------------------------------
extern "C" void kernel_launch(void* const* d_in, const int* in_sizes, int n_in,
                              void* d_out, int out_size, void* d_ws, size_t ws_size,
                              hipStream_t stream) {
    const float* point = (const float*)d_in[0];
    const float* feat  = (const float*)d_in[1];
    const int*   idx   = (const int*)d_in[2];
    const float* Wq = (const float*)d_in[3];  const float* bq = (const float*)d_in[4];
    const float* Wk = (const float*)d_in[5];  const float* bk = (const float*)d_in[6];
    const float* Wv = (const float*)d_in[7];  const float* bv = (const float*)d_in[8];
    const float* pW1 = (const float*)d_in[9];  const float* pb1 = (const float*)d_in[10];
    const float* pg1 = (const float*)d_in[11]; const float* pbe1 = (const float*)d_in[12];
    const float* pm1 = (const float*)d_in[13]; const float* pv1 = (const float*)d_in[14];
    const float* pW2 = (const float*)d_in[15]; const float* pb2 = (const float*)d_in[16];
    const float* wg1 = (const float*)d_in[17]; const float* wbe1 = (const float*)d_in[18];
    const float* wm1 = (const float*)d_in[19]; const float* wv1 = (const float*)d_in[20];
    const float* wW1 = (const float*)d_in[21]; const float* wb1 = (const float*)d_in[22];
    const float* wg2 = (const float*)d_in[23]; const float* wbe2 = (const float*)d_in[24];
    const float* wm2 = (const float*)d_in[25]; const float* wv2 = (const float*)d_in[26];
    const float* wW2 = (const float*)d_in[27]; const float* wb2 = (const float*)d_in[28];

    char* ws   = (char*)d_ws;
    float* fq  = (float*)(ws + OFF_FQ);
    f16*  fkh  = (f16*)(ws + OFF_FKH);
    f16*  fvh  = (f16*)(ws + OFF_FVH);
    float* bnp = (float*)(ws + OFF_BNP);
    f16*  bqkv = (f16*)(ws + OFF_BQKV);
    f16*  bw1p = (f16*)(ws + OFF_BW1);

    prep_kernel<<<4, 256, 0, stream>>>(Wq, Wk, Wv, pg1, pbe1, pm1, pv1,
                                       wg1, wbe1, wm1, wv1, wg2, wbe2, wm2, wv2,
                                       wW1, bnp, bqkv, bw1p);
    qkv_kernel<<<(NPTS / 16 + 7) / 8, 256, 0, stream>>>(feat, bq, bk, bv, bqkv, fq, fkh, fvh);
    attn_kernel<<<NPTS / 8, 256, 0, stream>>>(point, idx, fq, fkh, fvh,
                                              pW1, pb1, pW2, pb2, bnp, bw1p,
                                              wb1, wW2, wb2, (float*)d_out);
    (void)in_sizes; (void)n_in; (void)out_size; (void)ws_size;
}
